// MessageBlock_29480655519975
// MI455X (gfx1250) — matrix operand order repaired and checked by reading, not compiled, not run on hardware
//
#include <hip/hip_runtime.h>
#include <hip/hip_bf16.h>
#include <math.h>

// ---------------------------------------------------------------------------
// MessageBlock forward for MI455X (gfx1250), wave32 + bf16 WMMA.
// - B tiles stored transposed in LDS: every fragment = 2x ds_load_b128.
// - B staging fetches 4 ADJACENT k-rows per thread so transposed stores are
//   v4bf (ds_store_b64) instead of 16x ds_store_b16.
// - GEMM requires K%32==0 (feat padded to 2080): branch-free k-loop.
// - Register-staged pipeline + double-buffered LDS.
// ---------------------------------------------------------------------------

typedef __bf16 bf16;
typedef __attribute__((ext_vector_type(16))) __bf16 v16bf;
typedef __attribute__((ext_vector_type(8)))  __bf16 v8bf;
typedef __attribute__((ext_vector_type(4)))  __bf16 v4bf;
typedef __attribute__((ext_vector_type(8)))  float  v8f;

namespace cfg {
constexpr int N  = 2048;
constexpr int E  = 4096;
constexpr int L  = 49;
constexpr int LR = 16;
constexpr int M  = 25;
constexpr int NY = 3;
constexpr int C  = 128;
constexpr int H  = 128;
constexpr int NB = 128;
constexpr int FEAT  = LR * (C + 1);  // 2064 (real)
constexpr int FEATP = 2080;          // padded to multiple of 32
constexpr int LRC   = LR * C;        // 2048
}
#define INV_SQRT_3 0.57735026918962576f

// ---------------- WMMA helpers (ISA 7.12.2 layouts, wave32) ----------------

__device__ __forceinline__ v8f zero8() {
  v8f z;
#pragma unroll
  for (int i = 0; i < 8; ++i) z[i] = 0.f;
  return z;
}

__device__ __forceinline__ v8bf zero8bf() {
  v8bf z;
#pragma unroll
  for (int i = 0; i < 8; ++i) z[i] = (bf16)0.f;
  return z;
}

__device__ __forceinline__ v4bf zero4bf() {
  v4bf z;
#pragma unroll
  for (int i = 0; i < 4; ++i) z[i] = (bf16)0.f;
  return z;
}

__device__ __forceinline__ v8f wmma_bf16(v16bf a, v16bf b, v8f c) {
  return __builtin_amdgcn_wmma_f32_16x16x32_bf16(false, a, false, b, (short)0,
                                                 c, false, false);
}

// A fragment from row-major 16x32 bf16 tile (leading dim lda, lda%8==0).
__device__ __forceinline__ v16bf load_a_frag(const bf16* A, int lda) {
  const int lane = threadIdx.x & 31;
  const bf16* p = A + (lane & 15) * lda + ((lane >> 4) << 3);
  const v8bf lo = *(const v8bf*)(p);
  const v8bf hi = *(const v8bf*)(p + 16);
  return __builtin_shufflevector(lo, hi, 0, 1, 2, 3, 4, 5, 6, 7,
                                 8, 9, 10, 11, 12, 13, 14, 15);
}

// B fragment from TRANSPOSED tile BT[n][k] (leading dim ldk, ldk%8==0).
__device__ __forceinline__ v16bf load_b_frag(const bf16* BT, int ldk) {
  const int lane = threadIdx.x & 31;
  const bf16* p = BT + (lane & 15) * ldk + ((lane >> 4) << 3);
  const v8bf lo = *(const v8bf*)(p);
  const v8bf hi = *(const v8bf*)(p + 16);
  return __builtin_shufflevector(lo, hi, 0, 1, 2, 3, 4, 5, 6, 7,
                                 8, 9, 10, 11, 12, 13, 14, 15);
}

__device__ __forceinline__ float silu(float v) {
  return v * (1.f / (1.f + __expf(-v)));
}

// pack 4 K-adjacent values (one column) into a v4bf
#define PK4(a, b, c, d) v4bf{(bf16)(a), (bf16)(b), (bf16)(c), (bf16)(d)}

// ---------------- generic bf16-WMMA GEMM: D = act(A[M,K]@B[K,N] + bias) ----
// Requires: M%64==0, N%128==0, K%32==0.  Pipelined, double-buffered LDS.

#define GBM 64
#define GBN 128
#define GBK 32

__global__ __launch_bounds__(256) void gemm_bias_act(
    const float* __restrict__ A, const float* __restrict__ B,
    const float* __restrict__ bias, float* __restrict__ D,
    int M, int N, int K, int act) {
  __shared__ __align__(16) bf16 sA[2][GBM * GBK];   // row-major [m][k]
  __shared__ __align__(16) bf16 sBT[2][GBN * GBK];  // transposed [n][k]
  const int tid  = threadIdx.x;
  const int lane = tid & 31;
  const int wid  = tid >> 5;
  const int wm   = wid & 3;   // row tile 0..3
  const int wn   = wid >> 2;  // 64-col half 0..1
  const int row0 = blockIdx.y * GBM;
  const int col0 = blockIdx.x * GBN;

  // fixed per-thread staging coordinates
  const int arr = tid >> 3;          // A row base (0..31), +32 for chunk 1
  const int arc = (tid & 7) * 4;     // A k offset
  const int brg = (tid >> 5) * 4;    // B k row group: 4 adjacent rows
  const int brc = (tid & 31) * 4;    // B col offset

  v8f acc[4];
#pragma unroll
  for (int t = 0; t < 4; ++t) acc[t] = zero8();

  float4 ra[2], rb[4];
  // prefetch k-step 0
#pragma unroll
  for (int i = 0; i < 2; ++i)
    ra[i] = *(const float4*)(A + (size_t)(row0 + arr + 32 * i) * K + arc);
#pragma unroll
  for (int i = 0; i < 4; ++i)
    rb[i] = *(const float4*)(B + (size_t)(brg + i) * N + col0 + brc);

  const int nk = K / GBK;
  for (int kk = 0; kk < nk; ++kk) {
    const int buf = kk & 1;
    // stash prefetched chunks into LDS (bf16)
#pragma unroll
    for (int i = 0; i < 2; ++i) {
      *(v4bf*)(sA[buf] + (arr + 32 * i) * GBK + arc) =
          PK4(ra[i].x, ra[i].y, ra[i].z, ra[i].w);
    }
    // transposed B: one v4bf (8-byte) store per column
    *(v4bf*)(sBT[buf] + (brc + 0) * GBK + brg) =
        PK4(rb[0].x, rb[1].x, rb[2].x, rb[3].x);
    *(v4bf*)(sBT[buf] + (brc + 1) * GBK + brg) =
        PK4(rb[0].y, rb[1].y, rb[2].y, rb[3].y);
    *(v4bf*)(sBT[buf] + (brc + 2) * GBK + brg) =
        PK4(rb[0].z, rb[1].z, rb[2].z, rb[3].z);
    *(v4bf*)(sBT[buf] + (brc + 3) * GBK + brg) =
        PK4(rb[0].w, rb[1].w, rb[2].w, rb[3].w);
    __syncthreads();
    // prefetch next k-step while computing this one
    if (kk + 1 < nk) {
      const int k0 = (kk + 1) * GBK;
#pragma unroll
      for (int i = 0; i < 2; ++i)
        ra[i] = *(const float4*)(A + (size_t)(row0 + arr + 32 * i) * K +
                                 k0 + arc);
#pragma unroll
      for (int i = 0; i < 4; ++i)
        rb[i] = *(const float4*)(B + (size_t)(k0 + brg + i) * N +
                                 col0 + brc);
    }
    const v16bf a = load_a_frag(sA[buf] + wm * 16 * GBK, GBK);
    v16bf bf[4];
#pragma unroll
    for (int t = 0; t < 4; ++t)
      bf[t] = load_b_frag(sBT[buf] + (wn * 64 + t * 16) * GBK, GBK);
#pragma unroll
    for (int t = 0; t < 4; ++t) acc[t] = wmma_bf16(a, bf[t], acc[t]);
    __syncthreads();
  }

#pragma unroll
  for (int t = 0; t < 4; ++t) {
    const int n = col0 + wn * 64 + t * 16 + (lane & 15);
    const float bv = bias ? bias[n] : 0.f;
#pragma unroll
    for (int r = 0; r < 8; ++r) {
      const int m = row0 + wm * 16 + r + ((lane >> 4) << 3);
      float v = acc[t][r] + bv;
      if (act) v = silu(v);
      D[(size_t)m * N + n] = v;
    }
  }
}

// ---------------- stage 16 real k-rows transposed (ldk=32) -----------------
// threads 0..127: 4-row groups; threads 128..255: zero-pad rows 16..31.
// getv(r, c4) must return float4 of row r, cols c4..c4+3.

template <typename F>
__device__ __forceinline__ void stage16_transposed(bf16* sBT, int tid, F getv) {
  if (tid < 128) {
    const int g4 = (tid >> 5) * 4;  // 0,4,8,12
    const int c4 = (tid & 31) * 4;
    const float4 v0 = getv(g4 + 0, c4);
    const float4 v1 = getv(g4 + 1, c4);
    const float4 v2 = getv(g4 + 2, c4);
    const float4 v3 = getv(g4 + 3, c4);
    *(v4bf*)(sBT + (c4 + 0) * 32 + g4) = PK4(v0.x, v1.x, v2.x, v3.x);
    *(v4bf*)(sBT + (c4 + 1) * 32 + g4) = PK4(v0.y, v1.y, v2.y, v3.y);
    *(v4bf*)(sBT + (c4 + 2) * 32 + g4) = PK4(v0.z, v1.z, v2.z, v3.z);
    *(v4bf*)(sBT + (c4 + 3) * 32 + g4) = PK4(v0.w, v1.w, v2.w, v3.w);
  } else {
    const int c = tid - 128;
    *(v8bf*)(sBT + c * 32 + 16) = zero8bf();
    *(v8bf*)(sBT + c * 32 + 24) = zero8bf();
  }
}

// ---------------- pad weight rows with zeros: dst[rows_pad,ncol] -----------

__global__ void pad_rows_kernel(const float* __restrict__ src,
                                float* __restrict__ dst, int rows,
                                int rows_pad, int ncol) {
  const int idx = blockIdx.x * blockDim.x + threadIdx.x;
  if (idx >= rows_pad * ncol) return;
  dst[idx] = ((idx / ncol) < rows) ? src[idx] : 0.f;
}

// ---------------- per-node channel means: xmean[N,L] -----------------------

__global__ void node_mean_kernel(const float* __restrict__ x,
                                 float* __restrict__ xmean) {
  const int idx = blockIdx.x * blockDim.x + threadIdx.x;
  if (idx >= cfg::N * cfg::L) return;
  const float* p = x + (size_t)idx * cfg::C;
  float s = 0.f;
#pragma unroll 8
  for (int c = 0; c < cfg::C; ++c) s += p[c];
  xmean[idx] = s * (1.f / cfg::C);
}

// ---------------- CG bilinears: cg[E,L] ------------------------------------

__global__ __launch_bounds__(256) void cg_kernel(
    const float* __restrict__ xmean, const int* __restrict__ eidx,
    const float* __restrict__ W1, const float* __restrict__ W21,
    const float* __restrict__ W22, float* __restrict__ cg) {
  __shared__ float xm[cfg::L], ym[cfg::L], mid[cfg::M], cgs[cfg::L];
  const int e = blockIdx.x;
  const int tid = threadIdx.x;
  const int src = eidx[e], dst = eidx[cfg::E + e];
  if (tid < cfg::L) {
    xm[tid] = xmean[src * cfg::L + tid];
    ym[tid] = xmean[dst * cfg::L + tid];
    cgs[tid] = 0.f;
  }
  if (tid < cfg::M) mid[tid] = 0.f;
  __syncthreads();
  for (int p = tid; p < cfg::L * cfg::M; p += 256) {
    const int i = p / cfg::M, o = p % cfg::M;
    float s = 0.f;
    for (int j = 0; j < cfg::L; ++j)
      s += ym[j] * W1[(i * cfg::L + j) * cfg::M + o];
    atomicAdd(&mid[o], xm[i] * s);
  }
  __syncthreads();
  for (int p = tid; p < cfg::L * cfg::L; p += 256) {
    const int i = p / cfg::L, l = p % cfg::L;
    float s1 = 0.f, s2 = 0.f;
    for (int j = 0; j < cfg::M; ++j) {
      s1 += mid[j] * W21[(i * cfg::M + j) * cfg::L + l];
      s2 += mid[j] * W22[(i * cfg::M + j) * cfg::L + l];
    }
    atomicAdd(&cgs[l], xm[i] * s1 + ym[i] * s2);
  }
  __syncthreads();
  if (tid < cfg::L) cg[(size_t)e * cfg::L + tid] = cgs[tid];
}

// ---------------- node_int front: ne = wig^T @ x[node][:16], build feat ----

__global__ __launch_bounds__(256) void node_feat_kernel(
    const float* __restrict__ x, const float* __restrict__ glovec,
    const int* __restrict__ eidx, const float* __restrict__ wig_node,
    float* __restrict__ feat, int use_src) {
  __shared__ __align__(16) bf16 sA[16 * 32];    // wig^T padded K16->32
  __shared__ __align__(16) bf16 sBT[128 * 32];  // transposed [n][k]
  __shared__ float sne[16 * 128];
  __shared__ float smean[16];
  const int e = blockIdx.x;
  const int tid = threadIdx.x;
  const int lane = tid & 31;
  const int w = tid >> 5;
  const int node  = use_src ? eidx[e] : eidx[cfg::E + e];
  const int other = use_src ? eidx[cfg::E + e] : eidx[e];
  const float* wg = wig_node + (size_t)e * 256;
  for (int p = tid; p < 16 * 32; p += 256) {
    const int r = p >> 5, c = p & 31;
    sA[p] = (c < 16) ? (bf16)wg[c * 16 + r] : (bf16)0.f;  // transpose
  }
  const float* xp = x + (size_t)node * (cfg::L * cfg::C);
  stage16_transposed(sBT, tid, [&](int r, int c4) {
    return *(const float4*)(xp + r * cfg::C + c4);
  });
  __syncthreads();
  {
    const v16bf a = load_a_frag(sA, 32);
    const v16bf b = load_b_frag(sBT + (w * 16) * 32, 32);
    v8f acc = zero8();
    acc = wmma_bf16(a, b, acc);
    const int n = w * 16 + (lane & 15);
#pragma unroll
    for (int r = 0; r < 8; ++r)
      sne[(r + ((lane >> 4) << 3)) * 128 + n] = acc[r];
  }
  __syncthreads();
  if (tid < 16) {
    float s = 0.f;
    for (int c = 0; c < 128; ++c) s += sne[tid * 128 + c];
    smean[tid] = s * (1.f / 128.f);
  }
  __syncthreads();
  float* fp = feat + (size_t)e * cfg::FEATP;
  for (int p = tid; p < cfg::FEATP; p += 256) {
    float v = 0.f;
    if (p < cfg::FEAT) {
      const int l = p / (cfg::C + 1), c = p % (cfg::C + 1);
      v = (c < cfg::C) ? sne[l * 128 + c]
                       : smean[l] * glovec[other * cfg::LR + l];
    }
    fp[p] = v;
  }
}

// ---------------- node_int back: sh (+)= wig @ h ---------------------------

__global__ __launch_bounds__(256) void shc_kernel(
    const float* __restrict__ wig_node, const float* __restrict__ hh,
    float* __restrict__ sh, int accumulate) {
  __shared__ __align__(16) bf16 sA[16 * 32];
  __shared__ __align__(16) bf16 sBT[128 * 32];
  const int e = blockIdx.x;
  const int tid = threadIdx.x;
  const int lane = tid & 31;
  const int w = tid >> 5;
  const float* wg = wig_node + (size_t)e * 256;
  for (int p = tid; p < 16 * 32; p += 256) {
    const int r = p >> 5, c = p & 31;
    sA[p] = (c < 16) ? (bf16)wg[r * 16 + c] : (bf16)0.f;  // no transpose
  }
  const float* hp = hh + (size_t)e * cfg::LRC;
  stage16_transposed(sBT, tid, [&](int r, int c4) {
    return *(const float4*)(hp + r * 128 + c4);
  });
  __syncthreads();
  const v16bf a = load_a_frag(sA, 32);
  const v16bf b = load_b_frag(sBT + (w * 16) * 32, 32);
  v8f acc = zero8();
  acc = wmma_bf16(a, b, acc);
  float* sp = sh + (size_t)e * cfg::LRC;
  const int n = w * 16 + (lane & 15);
#pragma unroll
  for (int r = 0; r < 8; ++r) {
    const int m = r + ((lane >> 4) << 3);
    float v = acc[r];
    if (accumulate) v += sp[m * 128 + n];
    sp[m * 128 + n] = v;
  }
}

// ---------------- fused rotate: msg = wigner @ (2(xs+xt)+cg+pad(sh)) -------

__global__ __launch_bounds__(256) void rotate_kernel(
    const float* __restrict__ x, const int* __restrict__ eidx,
    const float* __restrict__ cg, const float* __restrict__ sh,
    const float* __restrict__ wigner, float* __restrict__ msg) {
  __shared__ __align__(16) bf16 sA[48 * 64];    // wigner [48,49] padded
  __shared__ __align__(16) bf16 sBT[128 * 64];  // transposed z [n][k]
  const int e = blockIdx.x;
  const int tid = threadIdx.x;
  const int lane = tid & 31;
  const int w = tid >> 5;
  const int src = eidx[e], dst = eidx[cfg::E + e];
  const float* xs = x + (size_t)src * (cfg::L * cfg::C);
  const float* xt = x + (size_t)dst * (cfg::L * cfg::C);
  const float* cgp = cg + (size_t)e * cfg::L;
  const float* shp = sh + (size_t)e * cfg::LRC;
  // rows 0..47 in 12 groups of 4 adjacent rows
  for (int p = tid; p < 12 * 32; p += 256) {
    const int g = p >> 5, c4 = (p & 31) * 4;
    const int r0 = g * 4;
    float4 z[4];
#pragma unroll
    for (int i = 0; i < 4; ++i) {
      const int r = r0 + i;
      const float4 a = *(const float4*)(xs + r * 128 + c4);
      const float4 b = *(const float4*)(xt + r * 128 + c4);
      const float cgv = cgp[r];
      z[i].x = 2.f * (a.x + b.x) + cgv;
      z[i].y = 2.f * (a.y + b.y) + cgv;
      z[i].z = 2.f * (a.z + b.z) + cgv;
      z[i].w = 2.f * (a.w + b.w) + cgv;
    }
    if (g < 4) {  // rows < 16 carry the sh contribution
#pragma unroll
      for (int i = 0; i < 4; ++i) {
        const float4 s = *(const float4*)(shp + (r0 + i) * 128 + c4);
        z[i].x += s.x; z[i].y += s.y; z[i].z += s.z; z[i].w += s.w;
      }
    }
    *(v4bf*)(sBT + (c4 + 0) * 64 + r0) = PK4(z[0].x, z[1].x, z[2].x, z[3].x);
    *(v4bf*)(sBT + (c4 + 1) * 64 + r0) = PK4(z[0].y, z[1].y, z[2].y, z[3].y);
    *(v4bf*)(sBT + (c4 + 2) * 64 + r0) = PK4(z[0].z, z[1].z, z[2].z, z[3].z);
    *(v4bf*)(sBT + (c4 + 3) * 64 + r0) = PK4(z[0].w, z[1].w, z[2].w, z[3].w);
  }
  // tail: row 48 + zero rows 49..63
  if (tid < 32) {
    const int c4 = tid * 4;
    const float4 a = *(const float4*)(xs + 48 * 128 + c4);
    const float4 b = *(const float4*)(xt + 48 * 128 + c4);
    const float cgv = cgp[48];
    float zv[4];
    zv[0] = 2.f * (a.x + b.x) + cgv;
    zv[1] = 2.f * (a.y + b.y) + cgv;
    zv[2] = 2.f * (a.z + b.z) + cgv;
    zv[3] = 2.f * (a.w + b.w) + cgv;
#pragma unroll
    for (int j = 0; j < 4; ++j) {
      bf16* col = sBT + (c4 + j) * 64;
      *(v4bf*)(col + 48) = PK4(zv[j], 0.f, 0.f, 0.f);
      *(v4bf*)(col + 52) = zero4bf();
      *(v8bf*)(col + 56) = zero8bf();
    }
  }
  const float* wp = wigner + (size_t)e * (48 * 49);
  for (int p = tid; p < 48 * 64; p += 256) {
    const int r = p >> 6, c = p & 63;
    sA[p] = (c < 49) ? (bf16)wp[r * 49 + c] : (bf16)0.f;
  }
  __syncthreads();
#pragma unroll
  for (int i = 0; i < 3; ++i) {  // 24 D tiles, 3 per wave
    const int tile = w + 8 * i;
    const int mt = tile >> 3, nt = tile & 7;
    v8f acc = zero8();
#pragma unroll
    for (int ks = 0; ks < 2; ++ks) {
      const v16bf a = load_a_frag(sA + mt * 16 * 64 + ks * 32, 64);
      const v16bf b = load_b_frag(sBT + (nt * 16) * 64 + ks * 32, 64);
      acc = wmma_bf16(a, b, acc);
    }
    float* mp = msg + ((size_t)e * cfg::NY + mt) * cfg::LRC;
    const int n = nt * 16 + (lane & 15);
#pragma unroll
    for (int r = 0; r < 8; ++r) {
      const int m = r + ((lane >> 4) << 3);
      mp[m * 128 + n] = acc[r];
    }
  }
}

// ---------------- broadcast scale by xe ------------------------------------

__global__ void scale_kernel(float* __restrict__ mm,
                             const float* __restrict__ xe) {
  const int idx = blockIdx.x * blockDim.x + threadIdx.x;
  if (idx >= cfg::E * cfg::NY * cfg::H) return;
  const int row = idx >> 7, h = idx & 127;
  mm[idx] *= xe[(row / cfg::NY) * cfg::H + h];
}

// ---------------- inverse rotation + NY mean -> out ------------------------

__global__ __launch_bounds__(256) void out_kernel(
    const float* __restrict__ m2, const float* __restrict__ wiginv,
    float* __restrict__ out) {
  __shared__ __align__(16) bf16 sA[64 * 32];    // wigner_inv [49,16] padded
  __shared__ __align__(16) bf16 sBT[128 * 32];  // transposed mred [n][k]
  const int e = blockIdx.x;
  const int tid = threadIdx.x;
  const int lane = tid & 31;
  const int w = tid >> 5;
  const float* mp = m2 + (size_t)e * cfg::NY * cfg::LRC;
  stage16_transposed(sBT, tid, [&](int r, int c4) {
    const int o = r * 128 + c4;
    const float4 a = *(const float4*)(mp + o);
    const float4 b = *(const float4*)(mp + cfg::LRC + o);
    const float4 c = *(const float4*)(mp + 2 * cfg::LRC + o);
    return make_float4((a.x + b.x + c.x) * (1.f / 3.f),
                       (a.y + b.y + c.y) * (1.f / 3.f),
                       (a.z + b.z + c.z) * (1.f / 3.f),
                       (a.w + b.w + c.w) * (1.f / 3.f));
  });
  const float* wvp = wiginv + (size_t)e * (cfg::L * cfg::LR);
  for (int p = tid; p < 64 * 32; p += 256) {
    const int r = p >> 5, c = p & 31;
    sA[p] = (r < cfg::L && c < cfg::LR) ? (bf16)wvp[r * cfg::LR + c]
                                        : (bf16)0.f;
  }
  __syncthreads();
  const v16bf b = load_b_frag(sBT + (w * 16) * 32, 32);
  const int n = w * 16 + (lane & 15);
#pragma unroll
  for (int mt = 0; mt < 4; ++mt) {
    const v16bf a = load_a_frag(sA + mt * 16 * 32, 32);
    v8f acc = zero8();
    acc = wmma_bf16(a, b, acc);
#pragma unroll
    for (int r = 0; r < 8; ++r) {
      const int l = mt * 16 + r + ((lane >> 4) << 3);
      if (l < cfg::L)
        out[((size_t)e * cfg::L + l) * cfg::C + n] = acc[r] * INV_SQRT_3;
    }
  }
}

// ---------------- host launcher --------------------------------------------

extern "C" void kernel_launch(void* const* d_in, const int* in_sizes, int n_in,
                              void* d_out, int out_size, void* d_ws,
                              size_t ws_size, hipStream_t stream) {
  (void)in_sizes; (void)n_in; (void)out_size; (void)ws_size;
  using namespace cfg;
  const float* x      = (const float*)d_in[0];
  const float* glovec = (const float*)d_in[2];
  const float* x_edge = (const float*)d_in[3];
  const int*   eidx   = (const int*)d_in[4];
  const float* W1   = (const float*)d_in[8];
  const float* W21  = (const float*)d_in[9];
  const float* W22  = (const float*)d_in[10];
  const float* Wn1a = (const float*)d_in[11];
  const float* bn1a = (const float*)d_in[12];
  const float* Wn1b = (const float*)d_in[13];
  const float* bn1b = (const float*)d_in[14];
  const float* Wn2a = (const float*)d_in[15];
  const float* bn2a = (const float*)d_in[16];
  const float* Wn2b = (const float*)d_in[17];
  const float* bn2b = (const float*)d_in[18];
  const float* Wd   = (const float*)d_in[19];
  const float* bd   = (const float*)d_in[20];
  const float* Wp1  = (const float*)d_in[21];
  const float* bp1  = (const float*)d_in[22];
  const float* Wp2  = (const float*)d_in[23];
  const float* bp2  = (const float*)d_in[24];
  const float* wigner = (const float*)d_in[25];
  const float* wiginv = (const float*)d_in[26];
  const float* wignode = (const float*)d_in[27];
  float* out = (float*)d_out;

  // workspace layout (floats), ~216 MB total
  float* ws = (float*)d_ws;
  float* xe     = ws;  ws += (size_t)E * H;          //  2.0 MB
  float* xmean  = ws;  ws += (size_t)N * L;          //  0.4 MB
  float* cg     = ws;  ws += (size_t)E * L;          //  0.8 MB
  float* sh     = ws;  ws += (size_t)E * LRC;        // 33.6 MB
  float* feat   = ws;  ws += (size_t)E * FEATP;      // 34.1 MB (reused x2)
  float* h      = ws;  ws += (size_t)E * H;          //  2.0 MB (reused x2)
  float* hh     = ws;  ws += (size_t)E * LRC;        // 33.6 MB (reused x2)
  float* msg    = ws;  ws += (size_t)E * NY * LRC;   // 100.7 MB (reused as m2)
  float* mm     = ws;  ws += (size_t)E * NY * H;     //  6.3 MB
  float* Wn1aP  = ws;  ws += (size_t)FEATP * H;      //  1.1 MB
  float* Wn2aP  = ws;  ws += (size_t)FEATP * H;      //  1.1 MB

  // 0) zero-pad node-MLP first-layer weights to K=2080 rows
  pad_rows_kernel<<<(FEATP * H + 255) / 256, 256, 0, stream>>>(
      Wn1a, Wn1aP, FEAT, FEATP, H);
  pad_rows_kernel<<<(FEATP * H + 255) / 256, 256, 0, stream>>>(
      Wn2a, Wn2aP, FEAT, FEATP, H);

  // 1) per-node channel means
  node_mean_kernel<<<(N * L + 255) / 256, 256, 0, stream>>>(x, xmean);
  // 2) xe = silu(x_edge @ Wd + bd)
  gemm_bias_act<<<dim3(H / GBN, E / GBM), 256, 0, stream>>>(
      x_edge, Wd, bd, xe, E, H, NB, 1);
  // 3) CG bilinears -> cg[E,L]
  cg_kernel<<<E, 256, 0, stream>>>(xmean, eidx, W1, W21, W22, cg);

  // 4) node_interaction1 (src features, dst glovec)
  node_feat_kernel<<<E, 256, 0, stream>>>(x, glovec, eidx, wignode, feat, 1);
  gemm_bias_act<<<dim3(H / GBN, E / GBM), 256, 0, stream>>>(
      feat, Wn1aP, bn1a, h, E, H, FEATP, 1);
  gemm_bias_act<<<dim3(LRC / GBN, E / GBM), 256, 0, stream>>>(
      h, Wn1b, bn1b, hh, E, LRC, H, 1);
  shc_kernel<<<E, 256, 0, stream>>>(wignode, hh, sh, 0);

  // 5) node_interaction2 (dst features, src glovec)
  node_feat_kernel<<<E, 256, 0, stream>>>(x, glovec, eidx, wignode, feat, 0);
  gemm_bias_act<<<dim3(H / GBN, E / GBM), 256, 0, stream>>>(
      feat, Wn2aP, bn2a, h, E, H, FEATP, 1);
  gemm_bias_act<<<dim3(LRC / GBN, E / GBM), 256, 0, stream>>>(
      h, Wn2b, bn2b, hh, E, LRC, H, 1);
  shc_kernel<<<E, 256, 0, stream>>>(wignode, hh, sh, 1);

  // 6) fused triple-rotation: msg = wigner @ (2(xs+xt) + cg + pad(sh))
  rotate_kernel<<<E, 256, 0, stream>>>(x, eidx, cg, sh, wigner, msg);

  // 7) mm = silu(msg @ Wp1 + bp1)
  gemm_bias_act<<<dim3(H / GBN, (E * NY) / GBM), 256, 0, stream>>>(
      msg, Wp1, bp1, mm, E * NY, H, LRC, 1);
  // 8) mm *= xe (broadcast over NY)
  scale_kernel<<<(E * NY * H + 255) / 256, 256, 0, stream>>>(mm, xe);
  // 9) m2 = silu(mm @ Wp2 + bp2)  (reuse msg buffer)
  gemm_bias_act<<<dim3(LRC / GBN, (E * NY) / GBM), 256, 0, stream>>>(
      mm, Wp2, bp2, msg, E * NY, LRC, H, 1);
  // 10) mean over NY + inverse rotation -> out
  out_kernel<<<E, 256, 0, stream>>>(msg, wiginv, out);
}
